// DynamicMoE_14499809592010
// MI455X (gfx1250) — matrix operand order repaired and checked by reading, not compiled, not run on hardware
//
#include <hip/hip_runtime.h>

// Problem constants (from the reference): B=2, S=2048, D=768, E=8, H=4*D=3072
#define TOK   4096
#define DIM   768
#define EXP   8
#define HID   3072
#define HC    128                 // H chunk (8 waves x 1 tile x 16)
#define MTILE 32                  // tokens per block (two WMMA M tiles)
#define MAX_TILES (TOK / MTILE)   // 128
#define XP    (DIM + 8)           // LDS pitch (bf16) for X tile
#define HP    (HC + 8)            // LDS pitch (bf16) for H tile
#define KT1   (DIM / 32)          // 24  k-tiles in GEMM1
#define NT1   (HID / 16)          // 192 n-tiles in GEMM1
#define KT2   (HID / 32)          // 96  k-tiles in GEMM2
#define NT2   (DIM / 16)          // 48  n-tiles in GEMM2

// workspace layout (bytes)
#define IDX_OFF  1024
#define W1P_OFF  (IDX_OFF + EXP * TOK * 4)                 // 132096
#define W2P_OFF  (W1P_OFF + (size_t)EXP * DIM * HID * 2)   // +37.7MB
#define WS_NEED  (W2P_OFF + (size_t)EXP * HID * DIM * 2)   // ~75.6MB

typedef __attribute__((ext_vector_type(16))) __bf16        v16bf;
typedef __attribute__((ext_vector_type(8)))  float         v8f;
typedef __attribute__((ext_vector_type(4)))  unsigned int  v4u;

union FragBF {
    v16bf          bf;
    unsigned short u[16];
    v4u            q[2];
};

__device__ __forceinline__ unsigned short f2bf(float f) {
    unsigned int u = __float_as_uint(f);
    u += 0x7FFFu + ((u >> 16) & 1u);
    return (unsigned short)(u >> 16);
}

// ---------------- kernel 1: zero per-expert counters ----------------
__global__ void moe_init_kernel(int* __restrict__ counts) {
    if (threadIdx.x < EXP) counts[threadIdx.x] = 0;
}

// ---------------- kernel 2: bucket token ids per expert -------------
__global__ void moe_route_kernel(const int* __restrict__ route,
                                 int* __restrict__ counts,
                                 int* __restrict__ idxbuf) {
    int t = blockIdx.x * blockDim.x + threadIdx.x;
    if (t >= TOK) return;
    int e = route[t] & (EXP - 1);
    int pos = atomicAdd(&counts[e], 1);
    idxbuf[e * TOK + pos] = t;
}

// ---- kernel 2.5: convert W (K x N, row-major f32) to bf16 in WMMA B-fragment order.
// Tile (kt,nt): K rows [kt*32,+32), N cols [nt*16,+16). Per lane: 16 contiguous bf16 =
// column (nt*16 + lane%16), K rows (kt*32 + 16*(lane/16) + 0..15).
__global__ __launch_bounds__(256)
void moe_pack_kernel(const float* __restrict__ src, unsigned short* __restrict__ dst,
                     int K, int N) {
    const int lane = threadIdx.x & 31;
    const int wid  = blockIdx.x * (blockDim.x >> 5) + (threadIdx.x >> 5);
    const int KT = K >> 5, NT = N >> 4;
    const int tilesPer = KT * NT;
    if (wid >= EXP * tilesPer) return;
    const int e  = wid / tilesPer;
    const int r  = wid % tilesPer;
    const int nt = r / KT;
    const int kt = r % KT;
    const float* s = src + (size_t)e * K * N
                   + (size_t)(kt * 32 + 16 * (lane >> 4)) * N + nt * 16 + (lane & 15);
    FragBF f;
    #pragma unroll
    for (int j = 0; j < 16; ++j) f.u[j] = f2bf(s[(size_t)j * N]);
    v4u* o = (v4u*)(dst + (size_t)e * K * N + ((((size_t)nt * KT + kt) * 32 + lane) << 4));
    o[0] = f.q[0];
    o[1] = f.q[1];
}

// ---------------- kernel 3: fused expert FFN over one 32-token tile -
template <bool PACKED>
__global__ __launch_bounds__(256)
void moe_ffn_kernel(const float* __restrict__ x,
                    const float* __restrict__ W1, const float* __restrict__ b1,
                    const float* __restrict__ W2, const float* __restrict__ b2,
                    const unsigned short* __restrict__ W1p,
                    const unsigned short* __restrict__ W2p,
                    const int* __restrict__ counts, const int* __restrict__ idxbuf,
                    float* __restrict__ out) {
    __shared__ int tokid[MTILE];
    __shared__ __align__(16) unsigned short Xt[MTILE * XP];   // 48.5 KB
    __shared__ __align__(16) unsigned short Ht[MTILE * HP];   //  8.5 KB

    const int e    = blockIdx.x / MAX_TILES;
    const int tile = blockIdx.x % MAX_TILES;
    const int cnt  = counts[e];
    if (tile * MTILE >= cnt) return;           // uniform early exit

    const int tid  = threadIdx.x;
    const int lane = tid & 31;                 // wave32
    const int wave = tid >> 5;                 // 0..7
    const int nl   = lane & 15;
    const int half = lane >> 4;

    if (tid < MTILE) {
        int r = tile * MTILE + tid;
        tokid[tid] = (r < cnt) ? idxbuf[e * TOK + r] : -1;
    }
    __syncthreads();

    for (int i = tid; i < MTILE * DIM; i += 256) {
        int m = i / DIM, k = i % DIM;
        int t = tokid[m];
        float v = (t >= 0) ? x[(size_t)t * DIM + k] : 0.0f;
        Xt[m * XP + k] = f2bf(v);
    }
    __syncthreads();

    const float* W1e = W1 + (size_t)e * DIM * HID;
    const float* W2e = W2 + (size_t)e * HID * DIM;
    const unsigned short* W1pe = W1p + (size_t)e * DIM * HID;
    const unsigned short* W2pe = W2p + (size_t)e * HID * DIM;

    v8f acc0[6] = {};                 // token rows 0..15, 96 output cols per wave
    v8f acc1[6] = {};                 // token rows 16..31
    const int dbase = wave * 96;

    for (int hb = 0; hb < HID; hb += HC) {
        // ---- Phase A: h chunk = relu(X @ W1[:, hb..hb+HC] + b1), 1 n-tile/wave, 2 M-tiles ----
        v8f hacc0 = {}, hacc1 = {};
        const int n0  = hb + wave * 16;
        const int ntA = (hb >> 4) + wave;      // packed n-tile index
        #pragma unroll 4
        for (int kk = 0; kk < DIM; kk += 32) {
            FragBF a0, a1, b;
            a0.q[0] = *(const v4u*)&Xt[nl * XP + kk + 16 * half];
            a0.q[1] = *(const v4u*)&Xt[nl * XP + kk + 16 * half + 8];
            a1.q[0] = *(const v4u*)&Xt[(16 + nl) * XP + kk + 16 * half];
            a1.q[1] = *(const v4u*)&Xt[(16 + nl) * XP + kk + 16 * half + 8];
            if constexpr (PACKED) {
                const int kt = kk >> 5;
                const v4u* pB = (const v4u*)(W1pe + ((((size_t)ntA * KT1 + kt) * 32 + lane) << 4));
                __builtin_prefetch((const char*)pB + 1024, 0, 3);   // next k tile
                b.q[0] = pB[0]; b.q[1] = pB[1];
            } else {
                const float* p0 = W1e + (size_t)(kk + 16 * half) * HID + n0 + nl;
                #pragma unroll
                for (int j = 0; j < 16; ++j) b.u[j] = f2bf(p0[(size_t)j * HID]);
            }
            // one B fragment feeds both M tiles
            hacc0 = __builtin_amdgcn_wmma_f32_16x16x32_bf16(false, a0.bf, false, b.bf,
                                                            (short)0, hacc0, false, false);
            hacc1 = __builtin_amdgcn_wmma_f32_16x16x32_bf16(false, a1.bf, false, b.bf,
                                                            (short)0, hacc1, false, false);
        }

        const float bias = b1[e * HID + n0 + nl];

        __syncthreads();   // all waves done reading previous Ht chunk
        #pragma unroll
        for (int v = 0; v < 8; ++v) {
            int m = v + 8 * half;              // C layout: lane=N, VGPR v -> M
            float h0 = fmaxf(hacc0[v] + bias, 0.0f);   // v_max_num_f32
            float h1 = fmaxf(hacc1[v] + bias, 0.0f);
            Ht[m * HP + wave * 16 + nl]        = f2bf(h0);
            Ht[(16 + m) * HP + wave * 16 + nl] = f2bf(h1);
        }
        __syncthreads();   // Ht chunk visible

        // ---- Phase B: acc += h_chunk @ W2[hb..hb+HC, dbase..dbase+96] ----
        #pragma unroll 2
        for (int kk = 0; kk < HC; kk += 32) {
            FragBF a0, a1;
            a0.q[0] = *(const v4u*)&Ht[nl * HP + kk + 16 * half];
            a0.q[1] = *(const v4u*)&Ht[nl * HP + kk + 16 * half + 8];
            a1.q[0] = *(const v4u*)&Ht[(16 + nl) * HP + kk + 16 * half];
            a1.q[1] = *(const v4u*)&Ht[(16 + nl) * HP + kk + 16 * half + 8];
            const int kt2 = (hb + kk) >> 5;
            #pragma unroll
            for (int dt = 0; dt < 6; ++dt) {
                FragBF b;
                if constexpr (PACKED) {
                    const v4u* pb = (const v4u*)(W2pe +
                        ((((size_t)(wave * 6 + dt) * KT2 + kt2) * 32 + lane) << 4));
                    __builtin_prefetch((const char*)pb + 1024, 0, 3);
                    b.q[0] = pb[0]; b.q[1] = pb[1];
                } else {
                    const float* pb = W2e + (size_t)(hb + kk + 16 * half) * DIM + dbase + nl;
                    #pragma unroll
                    for (int j = 0; j < 16; ++j)
                        b.u[j] = f2bf(pb[(size_t)j * DIM + dt * 16]);
                }
                acc0[dt] = __builtin_amdgcn_wmma_f32_16x16x32_bf16(false, a0.bf, false, b.bf,
                                                                   (short)0, acc0[dt], false, false);
                acc1[dt] = __builtin_amdgcn_wmma_f32_16x16x32_bf16(false, a1.bf, false, b.bf,
                                                                   (short)0, acc1[dt], false, false);
            }
        }
    }

    // ---- Scatter with b2 ----
    #pragma unroll
    for (int dt = 0; dt < 6; ++dt) {
        int col = dbase + dt * 16 + nl;
        float bias = b2[e * DIM + col];
        #pragma unroll
        for (int v = 0; v < 8; ++v) {
            int m = v + 8 * half;
            int t0 = tokid[m];
            int t1 = tokid[16 + m];
            if (t0 >= 0) out[(size_t)t0 * DIM + col] = acc0[dt][v] + bias;
            if (t1 >= 0) out[(size_t)t1 * DIM + col] = acc1[dt][v] + bias;
        }
    }
}

extern "C" void kernel_launch(void* const* d_in, const int* in_sizes, int n_in,
                              void* d_out, int out_size, void* d_ws, size_t ws_size,
                              hipStream_t stream) {
    const float* x     = (const float*)d_in[0];
    const int*   route = (const int*)d_in[1];
    const float* W1    = (const float*)d_in[2];
    const float* b1    = (const float*)d_in[3];
    const float* W2    = (const float*)d_in[4];
    const float* b2    = (const float*)d_in[5];
    float*       out   = (float*)d_out;

    int* counts = (int*)d_ws;
    int* idxbuf = (int*)((char*)d_ws + IDX_OFF);

    moe_init_kernel<<<1, 64, 0, stream>>>(counts);
    moe_route_kernel<<<TOK / 256, 256, 0, stream>>>(route, counts, idxbuf);

    if (ws_size >= WS_NEED) {
        unsigned short* W1pk = (unsigned short*)((char*)d_ws + W1P_OFF);
        unsigned short* W2pk = (unsigned short*)((char*)d_ws + W2P_OFF);
        const int waves = EXP * KT1 * NT1;          // == EXP*KT2*NT2 == 36864
        moe_pack_kernel<<<(waves + 7) / 8, 256, 0, stream>>>(W1, W1pk, DIM, HID);
        moe_pack_kernel<<<(waves + 7) / 8, 256, 0, stream>>>(W2, W2pk, HID, DIM);
        moe_ffn_kernel<true><<<EXP * MAX_TILES, 256, 0, stream>>>(
            x, W1, b1, W2, b2, W1pk, W2pk, counts, idxbuf, out);
    } else {
        moe_ffn_kernel<false><<<EXP * MAX_TILES, 256, 0, stream>>>(
            x, W1, b1, W2, b2, nullptr, nullptr, counts, idxbuf, out);
    }
}
